// MultiHeadSparseAttention_2_36386962932049
// MI455X (gfx1250) — matrix operand order repaired and checked by reading, not compiled, not run on hardware
//
#include <hip/hip_runtime.h>
#include <hip/hip_bf16.h>

// Problem constants (fixed by the reference).
#define E_DIM   1024
#define NHEADS  16
#define HDIM    64
#define LQ      2048
#define SKV     2048
#define NBATCH  2
#define MROWS   (LQ * NBATCH)   // 4096 rows for all projections
#define KSEL    1176            // int(2048 * sigmoid(0.3))

typedef __attribute__((ext_vector_type(16))) __bf16 v16bf;
typedef __attribute__((ext_vector_type(8)))  float  v8f;

union Frag  { uint4 u[2]; v16bf v; };
union BF16U { __bf16 b; unsigned short u; };
union FU    { unsigned u; float f; };

// ---------------------------------------------------------------------------
// Fragment loaders (CDNA5 wave32 WMMA VGPR layouts, ISA 7.12.2).
// A (16x32 bf16): lanes 0-15 -> M=lane, K = {0..7, 16..23}; lanes 16-31 -> K = {8..15, 24..31}
// B (32x16 bf16): lanes 0-15 -> N=lane, K = {0..7, 16..23}; lanes 16-31 -> K = {8..15, 24..31}
// ---------------------------------------------------------------------------
__device__ __forceinline__ v16bf load_a_frag(const __bf16* base, int ld, int row0, int kc) {
  const int lane = threadIdx.x & 31;
  const int hs   = lane >> 4;
  const __bf16* p = base + (size_t)(row0 + (lane & 15)) * ld + kc + hs * 8;
  Frag f;
  f.u[0] = *(const uint4*)(p);        // K chunk 0..7  (or 8..15)
  f.u[1] = *(const uint4*)(p + 16);   // K chunk 16..23 (or 24..31)
  return f.v;
}

__device__ __forceinline__ v16bf load_b_frag(const __bf16* base, int ld, int col0, int kc) {
  const int lane = threadIdx.x & 31;
  const int hs   = lane >> 4;
  const __bf16* p = base + (size_t)(col0 + (lane & 15)) * ld + kc + hs * 8;
  Frag f;
  f.u[0] = *(const uint4*)(p);
  f.u[1] = *(const uint4*)(p + 16);
  return f.v;
}

// ---------------------------------------------------------------------------
// CDNA5 async global -> LDS copies (tracked by ASYNCcnt, ISA 10 / 15.18.3).
// ---------------------------------------------------------------------------
__device__ __forceinline__ void async_b128(unsigned lds_off, const void* gaddr) {
  asm volatile("global_load_async_to_lds_b128 %0, %1, off"
               :: "v"(lds_off), "v"(gaddr) : "memory");
}
// Copy one contiguous 2 KB K tile (16 rows x 64 bf16): 4 x b128 per lane.
__device__ __forceinline__ void async_copy_2k(const void* gsrc, unsigned lds_off) {
  const int lane = threadIdx.x & 31;
#pragma unroll
  for (int i = 0; i < 4; ++i)
    async_b128(lds_off + lane * 16 + i * 512, (const char*)gsrc + lane * 16 + i * 512);
}
// Copy 16 rows x 64 bytes of V^T (row stride ldbytes): 2 x b128 per lane.
__device__ __forceinline__ void async_copy_v(const void* grow0, int ldbytes, unsigned lds_off) {
  const int lane = threadIdx.x & 31;
  const int row = lane & 15, half = lane >> 4;
  const char* g = (const char*)grow0 + (size_t)row * ldbytes + half * 32;
  const unsigned l = lds_off + row * 64 + half * 32;
  async_b128(l, g);
  async_b128(l + 16, g + 16);
}

// ---------------------------------------------------------------------------
// Wave32 reductions via lane permutes.
// ---------------------------------------------------------------------------
__device__ __forceinline__ int wave_sum_i(int v) {
#pragma unroll
  for (int off = 16; off > 0; off >>= 1) v += __shfl_xor(v, off, 32);
  return v;
}
__device__ __forceinline__ float wave_sum_f(float v) {
#pragma unroll
  for (int off = 16; off > 0; off >>= 1) v += __shfl_xor(v, off, 32);
  return v;
}
__device__ __forceinline__ int wave_max_i(int v) {
#pragma unroll
  for (int off = 16; off > 0; off >>= 1) {
    int o = __shfl_xor(v, off, 32);
    v = v > o ? v : o;
  }
  return v;
}

// Order-preserving bf16-bit <-> u16 key mapping.
__device__ __forceinline__ int map_key(unsigned short k) {
  return (k & 0x8000u) ? (int)((unsigned short)~k) : (int)(k | 0x8000u);
}
__device__ __forceinline__ float unmap_key(int m) {
  unsigned short k = (m & 0x8000) ? (unsigned short)(m & 0x7fff)
                                  : (unsigned short)(~(unsigned)m);
  FU fu; fu.u = ((unsigned)k) << 16;   // bf16 -> f32
  return fu.f;
}

// ---------------------------------------------------------------------------
// Kernel 1: fp32 -> bf16 conversion.
// ---------------------------------------------------------------------------
__global__ __launch_bounds__(256)
void cvt_f32_bf16(const float* __restrict__ in, __bf16* __restrict__ out, int n) {
  int i = blockIdx.x * 256 + threadIdx.x;
  if (i < n) out[i] = (__bf16)in[i];
}

// ---------------------------------------------------------------------------
// Kernel 2: GEMM  C[m,n] = sum_k A[m,k]*B[n,k] + bias[n]   (bf16 in, f32 acc)
// One wave computes a 64x64 tile (16 accumulators, 16 WMMA per K step).
// mode 0: f32 out, row-major [MROWS, E_DIM]          (output projection)
// mode 1: bf16 out scattered to [N,H,L,D]            (Q and K projections)
// mode 2: bf16 out scattered to [N,H,D,S] transposed (V projection)
// ---------------------------------------------------------------------------
__global__ __launch_bounds__(32)
void gemm_wmma_bf16(const __bf16* __restrict__ A, const __bf16* __restrict__ B,
                    const float* __restrict__ bias, void* __restrict__ outp, int mode) {
  const int m0 = blockIdx.x * 64;
  const int n0 = blockIdx.y * 64;
  const int lane = threadIdx.x & 31;
  const int hs = lane >> 4, nl = lane & 15;

  v8f acc[4][4] = {};
  for (int kc = 0; kc < E_DIM; kc += 32) {
    v16bf af[4], bf[4];
#pragma unroll
    for (int mi = 0; mi < 4; ++mi) af[mi] = load_a_frag(A, E_DIM, m0 + 16 * mi, kc);
#pragma unroll
    for (int ni = 0; ni < 4; ++ni) bf[ni] = load_b_frag(B, E_DIM, n0 + 16 * ni, kc);
#pragma unroll
    for (int mi = 0; mi < 4; ++mi)
#pragma unroll
      for (int ni = 0; ni < 4; ++ni)
        acc[mi][ni] = __builtin_amdgcn_wmma_f32_16x16x32_bf16(
            false, af[mi], false, bf[ni], (short)0, acc[mi][ni], false, false);
  }

#pragma unroll
  for (int mi = 0; mi < 4; ++mi)
#pragma unroll
    for (int ni = 0; ni < 4; ++ni)
#pragma unroll
      for (int r = 0; r < 8; ++r) {
        const int m  = m0 + 16 * mi + r + 8 * hs;   // C/D layout: M = r + 8*(lane>=16)
        const int nn = n0 + 16 * ni + nl;           // N = lane & 15
        const float vv = acc[mi][ni][r] + bias[nn];
        if (mode == 0) {
          ((float*)outp)[(size_t)m * E_DIM + nn] = vv;
        } else {
          const int l = m >> 1, nb = m & 1;         // m = l*NBATCH + nb
          const int h = nn >> 6, d = nn & 63;       // nn = h*HDIM + d
          size_t idx;
          if (mode == 1) idx = ((size_t)(nb * NHEADS + h) * LQ + l) * HDIM + d;
          else           idx = ((size_t)(nb * NHEADS + h) * HDIM + d) * SKV + l;
          ((__bf16*)outp)[idx] = (__bf16)vv;
        }
      }
}

// ---------------------------------------------------------------------------
// Kernel 3: fused attention per (batch, head, 16-query tile). One wave/block.
//   K/V tiles double-buffered into LDS with async global->LDS copies,
//   scores (bf16, scaled) -> LDS -> register top-1176 threshold + softmax
//   -> PV WMMA -> bf16 out with e = d*H + h ordering.
// ---------------------------------------------------------------------------
__global__ __launch_bounds__(32)
void attn_topk_wmma(const __bf16* __restrict__ q, const __bf16* __restrict__ k,
                    const __bf16* __restrict__ vt, __bf16* __restrict__ aout) {
  __shared__ unsigned short sh[16 * SKV];                       // 64 KB scores -> probs
  __shared__ __align__(16) unsigned short kbuf[2][16 * HDIM];   // 2 x 2 KB K tile ring
  __shared__ __align__(16) unsigned short vbuf[2][16 * 32];     // 2 x 1 KB V chunk ring

  const int qt = blockIdx.x;       // query tile (L/16)
  const int h  = blockIdx.y;       // head
  const int nb = blockIdx.z;       // batch
  const int lane = threadIdx.x & 31;
  const int hs = lane >> 4, nl = lane & 15;

  const __bf16* qbase = q  + (size_t)(nb * NHEADS + h) * LQ  * HDIM;
  const __bf16* kbase = k  + (size_t)(nb * NHEADS + h) * SKV * HDIM;
  const __bf16* vbase = vt + (size_t)(nb * NHEADS + h) * HDIM * SKV;

  const unsigned kbuf_lo[2] = {(unsigned)(size_t)&kbuf[0][0], (unsigned)(size_t)&kbuf[1][0]};
  const unsigned vbuf_lo[2] = {(unsigned)(size_t)&vbuf[0][0], (unsigned)(size_t)&vbuf[1][0]};

  // Q tile A-fragments (16 x 64, two K chunks), reused across all 128 S tiles.
  const v16bf aq0 = load_a_frag(qbase, HDIM, qt * 16, 0);
  const v16bf aq1 = load_a_frag(qbase, HDIM, qt * 16, 32);

  // ---- S = Q K^T / sqrt(D): async-staged K tiles, 16x16 score tiles to LDS ----
  async_copy_2k(kbase, kbuf_lo[0]);                       // prologue: tile 0
  for (int st = 0; st < SKV / 16; ++st) {
    if (st + 1 < SKV / 16) {
      async_copy_2k(kbase + (size_t)(st + 1) * 16 * HDIM, kbuf_lo[(st + 1) & 1]);
      asm volatile("s_wait_asynccnt 0x4" ::: "memory");   // tile st landed (in-order)
    } else {
      asm volatile("s_wait_asynccnt 0x0" ::: "memory");
    }
    const __bf16* kt = (const __bf16*)&kbuf[st & 1][0];
    const v16bf bk0 = load_b_frag(kt, HDIM, 0, 0);
    const v16bf bk1 = load_b_frag(kt, HDIM, 0, 32);
    v8f acc = {};
    acc = __builtin_amdgcn_wmma_f32_16x16x32_bf16(false, aq0, false, bk0, (short)0, acc, false, false);
    acc = __builtin_amdgcn_wmma_f32_16x16x32_bf16(false, aq1, false, bk1, (short)0, acc, false, false);
#pragma unroll
    for (int r = 0; r < 8; ++r) {
      const int qr = r + 8 * hs;
      const int sc = st * 16 + nl;
      BF16U bu; bu.b = (__bf16)(acc[r] * 0.125f);         // 1/sqrt(64)
      sh[qr * SKV + sc] = bu.u;
    }
  }
  __syncthreads();

  // ---- per-row top-KSEL threshold (binary search on order-mapped u16 keys) + softmax ----
  for (int r = 0; r < 16; ++r) {
    const int base = r * SKV + lane * 64;                 // each lane owns 64 contiguous keys
    int mk[64];
    const uint4* sp = (const uint4*)&sh[base];
#pragma unroll
    for (int w = 0; w < 8; ++w) {
      const uint4 q4 = sp[w];
      const unsigned wd[4] = {q4.x, q4.y, q4.z, q4.w};
#pragma unroll
      for (int d = 0; d < 4; ++d) {
        mk[w * 8 + d * 2 + 0] = map_key((unsigned short)(wd[d] & 0xffffu));
        mk[w * 8 + d * 2 + 1] = map_key((unsigned short)(wd[d] >> 16));
      }
    }
    // largest t with count(key >= t) >= KSEL  ==  KSEL-th largest key
    int thr = 0;
#pragma unroll
    for (int bit = 15; bit >= 0; --bit) {
      const int cand = thr | (1 << bit);
      int c = 0;
#pragma unroll
      for (int j = 0; j < 64; ++j) c += (mk[j] >= cand);
      if (wave_sum_i(c) >= KSEL) thr = cand;
    }
    int mx = 0;
#pragma unroll
    for (int j = 0; j < 64; ++j) mx = mk[j] > mx ? mk[j] : mx;
    const float vmax = unmap_key(wave_max_i(mx));

    float ssum = 0.f;
#pragma unroll
    for (int j = 0; j < 64; ++j)
      if (mk[j] >= thr) ssum += __expf(unmap_key(mk[j]) - vmax);
    const float inv = 1.0f / wave_sum_f(ssum);

#pragma unroll
    for (int j = 0; j < 64; ++j) {
      const float p = (mk[j] >= thr) ? __expf(unmap_key(mk[j]) - vmax) * inv : 0.f;
      BF16U bu; bu.b = (__bf16)p;
      sh[base + j] = bu.u;                                // probs overwrite keys in place
    }
  }
  __syncthreads();

  // ---- O = P V : attn (16 x 2048 bf16 in LDS) x async-staged V^T chunks ----
  const __bf16* pb = (const __bf16*)sh;
#pragma unroll 1
  for (int dt = 0; dt < 4; ++dt) {
    const __bf16* vrow0 = vbase + (size_t)dt * 16 * SKV;  // rows d = dt*16 .. dt*16+15
    v8f acc = {};
    async_copy_v(vrow0, SKV * 2, vbuf_lo[0]);             // prologue: chunk kc=0
    for (int kc = 0; kc < SKV; kc += 32) {
      if (kc + 32 < SKV) {
        async_copy_v((const char*)vrow0 + (kc + 32) * 2, SKV * 2, vbuf_lo[((kc >> 5) + 1) & 1]);
        asm volatile("s_wait_asynccnt 0x2" ::: "memory"); // chunk kc landed
      } else {
        asm volatile("s_wait_asynccnt 0x0" ::: "memory");
      }
      const v16bf ap = load_a_frag(pb, SKV, 0, kc);                          // probs from LDS
      const v16bf bv = load_b_frag((const __bf16*)&vbuf[(kc >> 5) & 1][0], 32, 0, 0);
      acc = __builtin_amdgcn_wmma_f32_16x16x32_bf16(false, ap, false, bv, (short)0, acc, false, false);
    }
#pragma unroll
    for (int r = 0; r < 8; ++r) {
      const int l  = qt * 16 + r + 8 * hs;
      const size_t m = (size_t)l * NBATCH + nb;
      const int dg = dt * 16 + nl;
      const int e  = dg * NHEADS + h;                     // reference's d-major, h-minor ordering
      aout[m * E_DIM + e] = (__bf16)acc[r];
    }
  }
}

// ---------------------------------------------------------------------------
// Host launcher.
// ---------------------------------------------------------------------------
extern "C" void kernel_launch(void* const* d_in, const int* in_sizes, int n_in,
                              void* d_out, int out_size, void* d_ws, size_t ws_size,
                              hipStream_t stream) {
  (void)in_sizes; (void)n_in; (void)out_size; (void)ws_size;
  const float* query = (const float*)d_in[0];
  const float* key   = (const float*)d_in[1];
  const float* value = (const float*)d_in[2];
  const float* Wq = (const float*)d_in[3];  const float* bq = (const float*)d_in[4];
  const float* Wk = (const float*)d_in[5];  const float* bk = (const float*)d_in[6];
  const float* Wv = (const float*)d_in[7];  const float* bv = (const float*)d_in[8];
  const float* Wo = (const float*)d_in[9];  const float* bo = (const float*)d_in[10];

  char* ws = (char*)d_ws;
  const size_t MB = 1u << 20;
  __bf16* xq  = (__bf16*)(ws + 0 * MB);    // [4096,1024] bf16 activations
  __bf16* xk  = (__bf16*)(ws + 8 * MB);
  __bf16* xv  = (__bf16*)(ws + 16 * MB);
  __bf16* wqb = (__bf16*)(ws + 24 * MB);   // [1024,1024] bf16 weights (row = e_out)
  __bf16* wkb = (__bf16*)(ws + 26 * MB);
  __bf16* wvb = (__bf16*)(ws + 28 * MB);
  __bf16* wob = (__bf16*)(ws + 30 * MB);
  __bf16* qb  = (__bf16*)(ws + 32 * MB);   // [N,H,L,D]
  __bf16* kb  = (__bf16*)(ws + 40 * MB);   // [N,H,S,D]
  __bf16* vtb = (__bf16*)(ws + 48 * MB);   // [N,H,D,S]
  __bf16* ao  = (__bf16*)(ws + 56 * MB);   // [4096,1024] attention output

  const int nAct = MROWS * E_DIM;          // 4,194,304
  const int nW   = E_DIM * E_DIM;          // 1,048,576
  cvt_f32_bf16<<<(nAct + 255) / 256, 256, 0, stream>>>(query, xq, nAct);
  cvt_f32_bf16<<<(nAct + 255) / 256, 256, 0, stream>>>(key,   xk, nAct);
  cvt_f32_bf16<<<(nAct + 255) / 256, 256, 0, stream>>>(value, xv, nAct);
  cvt_f32_bf16<<<(nW + 255) / 256, 256, 0, stream>>>(Wq, wqb, nW);
  cvt_f32_bf16<<<(nW + 255) / 256, 256, 0, stream>>>(Wk, wkb, nW);
  cvt_f32_bf16<<<(nW + 255) / 256, 256, 0, stream>>>(Wv, wvb, nW);
  cvt_f32_bf16<<<(nW + 255) / 256, 256, 0, stream>>>(Wo, wob, nW);

  dim3 ggrid(MROWS / 64, E_DIM / 64);      // (64, 16)
  gemm_wmma_bf16<<<ggrid, 32, 0, stream>>>(xq, wqb, bq, (void*)qb, 1);
  gemm_wmma_bf16<<<ggrid, 32, 0, stream>>>(xk, wkb, bk, (void*)kb, 1);
  gemm_wmma_bf16<<<ggrid, 32, 0, stream>>>(xv, wvb, bv, (void*)vtb, 2);

  dim3 agrid(LQ / 16, NHEADS, NBATCH);     // (128, 16, 2)
  attn_topk_wmma<<<agrid, 32, 0, stream>>>(qb, kb, vtb, ao);

  gemm_wmma_bf16<<<ggrid, 32, 0, stream>>>(ao, wob, bo, d_out, 0);
}
